// DenseTopKSAE_44160853737878
// MI455X (gfx1250) — matrix operand order repaired and checked by reading, not compiled, not run on hardware
//
#include <hip/hip_runtime.h>

// ---------------------------------------------------------------------------
// TopK SAE forward for MI455X (gfx1250, wave32, WMMA).
//   B=1024, R=8, C=1024, D=16384, K=32
// Encoder GEMM in BF16 WMMA (dominant cost), register-resident top-k,
// sparse (K=32) decoder gather.
// ---------------------------------------------------------------------------

namespace {
constexpr int kB = 1024;
constexpr int kR = 8;
constexpr int kC = 1024;
constexpr int kD = 16384;
constexpr int kK = 32;   // top-k

// encoder GEMM tiling
constexpr int TBM  = 128;        // batch rows per block
constexpr int TBN  = 128;        // feature (d) cols per block
constexpr int TBK  = 32;         // K-step (matches wmma 16x16x32)
constexpr int LPAD = 8;          // halves of padding per LDS row
constexpr int LSTR = TBK + LPAD; // LDS row stride in halves (40 -> 80B, 16B aligned)
}

typedef __attribute__((ext_vector_type(16))) __bf16 bf16x16;
typedef __attribute__((ext_vector_type(8)))  __bf16 bf16x8;
typedef __attribute__((ext_vector_type(8)))  float  floatx8;

// ---------------------------------------------------------------------------
// Kernel 1: h[b*R+r, d] = sum_c (x[b,r,c] - dec_b[r,c]) * enc_w[r,d,c] + enc_b[r,d]
// NT GEMM: both A rows (x) and B rows (enc_w[d,:]) are contiguous along K.
// ---------------------------------------------------------------------------
__global__ __launch_bounds__(256)
void sae_encode_wmma(const float* __restrict__ x,
                     const float* __restrict__ enc_w,
                     const float* __restrict__ enc_b,
                     const float* __restrict__ dec_b,
                     float* __restrict__ h)
{
    __shared__ __align__(16) __bf16 As[TBM * LSTR];
    __shared__ __align__(16) __bf16 Bs[TBN * LSTR];

    const int tid = threadIdx.x;
    const int b0  = blockIdx.x * TBM;   // blockIdx.x fastest -> 8 M-tiles share B panel in L2
    const int d0  = blockIdx.y * TBN;
    const int r   = blockIdx.z;

    // global->LDS loader mapping: 32 rows x 8 float4-columns per pass, 4 passes
    const int lrow = tid >> 3;          // 0..31
    const int lcol = (tid & 7) * 4;     // 0,4,...,28

    // wave / lane mapping: 8 waves as 2 (M) x 4 (N)
    const int lane = tid & 31;
    const int wid  = tid >> 5;
    const int wm   = wid >> 2;          // 0..1 -> 64 rows each
    const int wn   = wid & 3;           // 0..3 -> 32 cols each
    const int lm   = lane & 15;
    const int lhi  = lane >> 4;
    const int lk8  = lhi * 8;           // A-fragment K offset (halves)
    const int lk16 = lhi * 16;          // unused; B fragments also use lk8

    floatx8 acc[4][2];
#pragma unroll
    for (int mt = 0; mt < 4; ++mt)
#pragma unroll
        for (int nt = 0; nt < 2; ++nt)
            acc[mt][nt] = (floatx8)(0.0f);

    for (int k0 = 0; k0 < kC; k0 += TBK) {
        // pre-bias values for this K slice (same for all 128 batch rows)
        const float4 dbv = *(const float4*)(dec_b + (size_t)r * kC + k0 + lcol);

#pragma unroll
        for (int i = 0; i < 4; ++i) {
            const int row = lrow + 32 * i;
            // A: x[b0+row, r, k0+lcol..+3] minus decoder pre-bias
            const float4 xa = *(const float4*)(x + ((size_t)(b0 + row) * kR + r) * kC + k0 + lcol);
            __bf16* da = &As[row * LSTR + lcol];
            da[0] = (__bf16)(xa.x - dbv.x);
            da[1] = (__bf16)(xa.y - dbv.y);
            da[2] = (__bf16)(xa.z - dbv.z);
            da[3] = (__bf16)(xa.w - dbv.w);
            // B: enc_w[r, d0+row, k0+lcol..+3]
            const float4 wb = *(const float4*)(enc_w + ((size_t)r * kD + (d0 + row)) * kC + k0 + lcol);
            __bf16* db = &Bs[row * LSTR + lcol];
            db[0] = (__bf16)wb.x;
            db[1] = (__bf16)wb.y;
            db[2] = (__bf16)wb.z;
            db[3] = (__bf16)wb.w;
        }
        __syncthreads();

        // A fragments: lane holds row lm, halves K = lk8..lk8+7 and lk8+16..lk8+23
        bf16x16 af[4];
#pragma unroll
        for (int mt = 0; mt < 4; ++mt) {
            const __bf16* p = &As[(wm * 64 + mt * 16 + lm) * LSTR + lk8];
            bf16x8 lo = *(const bf16x8*)p;
            bf16x8 hi = *(const bf16x8*)(p + 16);
            af[mt] = __builtin_shufflevector(lo, hi, 0,1,2,3,4,5,6,7,8,9,10,11,12,13,14,15);
        }
        // B fragments: lane holds column lm, halves K = lk8..lk8+7 and lk8+16..lk8+23
        bf16x16 bfr[2];
#pragma unroll
        for (int nt = 0; nt < 2; ++nt) {
            const __bf16* p = &Bs[(wn * 32 + nt * 16 + lm) * LSTR + lk8];
            bf16x8 lo = *(const bf16x8*)p;
            bf16x8 hi = *(const bf16x8*)(p + 16);
            bfr[nt] = __builtin_shufflevector(lo, hi, 0,1,2,3,4,5,6,7,8,9,10,11,12,13,14,15);
        }

#pragma unroll
        for (int mt = 0; mt < 4; ++mt)
#pragma unroll
            for (int nt = 0; nt < 2; ++nt)
                acc[mt][nt] = __builtin_amdgcn_wmma_f32_16x16x32_bf16(
                    false, af[mt], false, bfr[nt], (short)0, acc[mt][nt], false, false);

        __syncthreads();
    }

    // Epilogue: C/D layout -> element (M = v + 8*lhi, N = lm) in VGPR v.
#pragma unroll
    for (int nt = 0; nt < 2; ++nt) {
        const int n  = d0 + wn * 32 + nt * 16 + lm;
        const float eb = enc_b[(size_t)r * kD + n];
#pragma unroll
        for (int mt = 0; mt < 4; ++mt) {
            const int mbase = b0 + wm * 64 + mt * 16 + lhi * 8;
#pragma unroll
            for (int v = 0; v < 8; ++v) {
                h[((size_t)(mbase + v) * kR + r) * kD + n] = acc[mt][nt][v] + eb;
            }
        }
    }
}

// ---------------------------------------------------------------------------
// Kernel 2: top-32 per row of h (row = b*R+r, length D=16384).
// Each of 256 threads keeps its 64 strided values in registers; 32 rounds of
// wave-shuffle + cross-wave argmax; winner is invalidated with a fully
// unrolled compare (no dynamic register indexing -> no scratch).
// ---------------------------------------------------------------------------
__global__ __launch_bounds__(256)
void sae_topk(const float* __restrict__ h,
              int* __restrict__ oidx,
              float* __restrict__ oval)
{
    const int row = blockIdx.x;            // b*R + r
    const int tid = threadIdx.x;
    const float* hrow = h + (size_t)row * kD;

    float v[64];
#pragma unroll
    for (int j = 0; j < 64; ++j) v[j] = hrow[j * 256 + tid];

    __shared__ float wv[8];
    __shared__ int   wi[8];
    __shared__ int   s_gi;

    const float NEG = -3.4e38f;

    for (int it = 0; it < kK; ++it) {
        float bv = NEG;
        int   bj = 0;
#pragma unroll
        for (int j = 0; j < 64; ++j) {
            if (v[j] > bv) { bv = v[j]; bj = j; }
        }
        int bi = bj * 256 + tid;

#pragma unroll
        for (int off = 16; off > 0; off >>= 1) {
            float ov = __shfl_down(bv, off, 32);
            int   oi = __shfl_down(bi, off, 32);
            if (ov > bv) { bv = ov; bi = oi; }
        }
        if ((tid & 31) == 0) { wv[tid >> 5] = bv; wi[tid >> 5] = bi; }
        __syncthreads();
        if (tid == 0) {
            float gv = wv[0]; int gi = wi[0];
#pragma unroll
            for (int w = 1; w < 8; ++w)
                if (wv[w] > gv) { gv = wv[w]; gi = wi[w]; }
            oidx[row * kK + it] = gi;
            oval[row * kK + it] = gv > 0.0f ? gv : 0.0f;   // relu(masked h)
            s_gi = gi;
        }
        __syncthreads();
        const int gi = s_gi;
        if ((gi & 255) == tid) {
            const int slot = gi >> 8;
#pragma unroll
            for (int j = 0; j < 64; ++j)
                if (j == slot) v[j] = NEG;
        }
    }
}

// ---------------------------------------------------------------------------
// Kernel 3: sparse decode. out[b,r,c] = dec_b[r,c] + sum_j val_j * dec_w[r,c,idx_j]
// ---------------------------------------------------------------------------
__global__ __launch_bounds__(256)
void sae_decode(const float* __restrict__ dec_w,
                const float* __restrict__ dec_b,
                const int* __restrict__ oidx,
                const float* __restrict__ oval,
                float* __restrict__ out)
{
    const int row = blockIdx.x;          // b*R + r
    const int r   = row & (kR - 1);
    const int tid = threadIdx.x;

    __shared__ int   sidx[kK];
    __shared__ float sval[kK];
    if (tid < kK) {
        sidx[tid] = oidx[row * kK + tid];
        sval[tid] = oval[row * kK + tid];
    }
    __syncthreads();

#pragma unroll
    for (int cc = 0; cc < kC / 256; ++cc) {
        const int c = cc * 256 + tid;
        float acc = dec_b[(size_t)r * kC + c];
        const float* wrow = dec_w + ((size_t)r * kC + c) * kD;
#pragma unroll 8
        for (int j = 0; j < kK; ++j)
            acc += sval[j] * wrow[sidx[j]];
        out[(size_t)row * kC + c] = acc;
    }
}

// ---------------------------------------------------------------------------
extern "C" void kernel_launch(void* const* d_in, const int* in_sizes, int n_in,
                              void* d_out, int out_size, void* d_ws, size_t ws_size,
                              hipStream_t stream)
{
    (void)in_sizes; (void)n_in; (void)out_size; (void)ws_size;

    const float* x     = (const float*)d_in[0];
    const float* enc_w = (const float*)d_in[1];
    const float* enc_b = (const float*)d_in[2];
    const float* dec_w = (const float*)d_in[3];
    const float* dec_b = (const float*)d_in[4];
    // d_in[5] is k (==32, baked in as kK)
    float* out = (float*)d_out;

    char* ws = (char*)d_ws;
    float* h = (float*)ws;                                          // [B*R, D] fp32
    const size_t hBytes   = (size_t)kB * kR * kD * sizeof(float);   // 512 MB
    int*   oidx = (int*)(ws + hBytes);                              // [B*R, 32]
    float* oval = (float*)(ws + hBytes + (size_t)kB * kR * kK * sizeof(int));

    dim3 egrid(kB / TBM, kD / TBN, kR);   // (8, 128, 8); x fastest -> B-panel L2 reuse
    sae_encode_wmma<<<egrid, 256, 0, stream>>>(x, enc_w, enc_b, dec_b, h);
    sae_topk<<<dim3(kB * kR), 256, 0, stream>>>(h, oidx, oval);
    sae_decode<<<dim3(kB * kR), 256, 0, stream>>>(dec_w, dec_b, oidx, oval, out);
}
